// SincKANLayer_18416819765818
// MI455X (gfx1250) — matrix operand order repaired and checked by reading, not compiled, not run on hardware
//
#include <hip/hip_runtime.h>

// SincKAN forward as a fused sinc-basis + f16 WMMA GEMM for gfx1250.
//   y[b,o] = sum_{i,d} sinc(pi*(x[b,i]*inv_h[s]+k[j]+eps)) * coeffs[i,o,d]
// M=4096, N=512, K=512*40=20480.
// Double-buffered LDS: sinc/cvt VALU for chunk t+1 co-executes with the
// XDL WMMA stream for chunk t (CDNA5 tracks f16 WMMA as TRANS ops).

#define IN_DIM   512
#define OUT_DIM  512
#define DB       40          // 5 scales * 8 shifts
#define BM       64          // rows (batch) per block
#define BN       128         // cols (out)  per block
#define IC       4           // i-values per K chunk
#define KC       (IC * DB)   // 160 = 5 * 32
#define KP       (KC + 8)    // padded row stride (168 halfs) -> conflict-free ds_load_b128
#define NT       256         // 8 waves (wave32)
#define NITER    (IN_DIM / IC)   // 128 K chunks

#define CSCALE     4096.0f       // exact pow2: lifts coeffs out of f16 subnormal range
#define CSCALE_INV (1.0f / 4096.0f)

typedef __attribute__((ext_vector_type(16))) _Float16     v16h;
typedef __attribute__((ext_vector_type(8)))  float        v8f;
typedef __attribute__((ext_vector_type(4)))  unsigned int u32x4;

union Frag16 { u32x4 u[2]; v16h h; };          // trivial types only (union-safe)
union Pack8  { u32x4 u;    _Float16 h[8]; };

// Fill one (As, Bs) ping-pong buffer for K chunk starting at input index i0.
__device__ __forceinline__ void fill_tiles(const float* __restrict__ x,
                                           const float* __restrict__ coeffs,
                                           _Float16* __restrict__ As,
                                           _Float16* __restrict__ Bs,
                                           int row0, int col0, int i0, int tid)
{
    // ---- A: sinc basis tile (BM x KC), computed on the fly -> f16
    {
        const int row = tid >> 2;          // BM*IC == NT: one (row, ii) per thread
        const int ii  = tid & 3;
        const float xv = x[(size_t)(row0 + row) * IN_DIM + (i0 + ii)];
        _Float16* dst = &As[row * KP + ii * DB];
        const float invh[5] = {1.f, 5.f, 9.f, 13.f, 17.f};
        #pragma unroll
        for (int s = 0; s < 5; ++s) {
            const float base = xv * invh[s];
            Pack8 pk;
            #pragma unroll
            for (int j = 0; j < 8; ++j) {
                const float r = base + (float)(j - 3) + 1e-20f;   // k = -3..4
                const float z = 3.14159265358979323846f * r;
                pk.h[j] = (_Float16)(__sinf(z) * __builtin_amdgcn_rcpf(z));
            }
            *(u32x4*)(dst + s * 8) = pk.u;   // 16B-aligned LDS store
        }
    }
    // ---- B: coeff tile (BN x KC), f32 -> scaled f16, stored [col][k]
    {
        const float* cbase = coeffs + (size_t)i0 * OUT_DIM * DB;
        for (int q = tid; q < BN * IC; q += NT) {    // 2 iterations
            const int o  = q >> 2;
            const int ii = q & 3;
            const float* src = cbase + ((size_t)ii * OUT_DIM + (col0 + o)) * DB;
            if (i0 + IC < IN_DIM)                     // prefetch next K chunk
                __builtin_prefetch(src + (size_t)IC * OUT_DIM * DB, 0, 3);
            _Float16* dst = &Bs[o * KP + ii * DB];
            #pragma unroll
            for (int v = 0; v < 5; ++v) {
                const float4 f0 = ((const float4*)src)[2 * v];
                const float4 f1 = ((const float4*)src)[2 * v + 1];
                Pack8 pk;
                pk.h[0] = (_Float16)(f0.x * CSCALE);
                pk.h[1] = (_Float16)(f0.y * CSCALE);
                pk.h[2] = (_Float16)(f0.z * CSCALE);
                pk.h[3] = (_Float16)(f0.w * CSCALE);
                pk.h[4] = (_Float16)(f1.x * CSCALE);
                pk.h[5] = (_Float16)(f1.y * CSCALE);
                pk.h[6] = (_Float16)(f1.z * CSCALE);
                pk.h[7] = (_Float16)(f1.w * CSCALE);
                *(u32x4*)(dst + v * 8) = pk.u;
            }
        }
    }
}

__launch_bounds__(NT, 1)
__global__ void sinckan_wmma(const float* __restrict__ x,
                             const float* __restrict__ coeffs,
                             float* __restrict__ out)
{
    // Ping-pong buffers: 2 * (21504 + 43008) = 129024 B LDS (gfx1250 WGP: 320 KB).
    __shared__ __align__(16) _Float16 AsBuf[2][BM * KP];
    __shared__ __align__(16) _Float16 BsBuf[2][BN * KP];

    const int tid  = threadIdx.x;
    const int lane = tid & 31;
    const int wid  = tid >> 5;
    const int wm   = wid >> 2;      // 0..1 : 32-row slab of the 64-row tile
    const int wn   = wid & 3;       // 0..3 : 32-col slab of the 128-col tile
    const int l15  = lane & 15;
    const int lhi  = lane >> 4;     // 0 | 1 (half-wave select per ISA layouts)

    const int row0 = blockIdx.y * BM;
    const int col0 = blockIdx.x * BN;

    v8f acc[2][2] = {};   // 2x2 16x16 f32 accumulator tiles per wave

    // Prologue: fill buffer 0 for chunk 0.
    fill_tiles(x, coeffs, AsBuf[0], BsBuf[0], row0, col0, 0, tid);
    __syncthreads();

    for (int it = 0; it < NITER; ++it) {
        const int buf = it & 1;

        // Fill the other buffer for the next chunk (overlaps with WMMA below).
        if (it + 1 < NITER)
            fill_tiles(x, coeffs, AsBuf[buf ^ 1], BsBuf[buf ^ 1],
                       row0, col0, (it + 1) * IC, tid);

        const _Float16* As = AsBuf[buf];
        const _Float16* Bs = BsBuf[buf];

        // ---- 5 WMMA k-steps of 32 over this chunk
        #pragma unroll
        for (int ks = 0; ks < KC; ks += 32) {
            Frag16 a[2], b[2];
            // A frags use contiguous k order chosen here (not ISA table order): lanes 0-15 hold K 0..15, lanes 16-31 K 16..31 (matches B)
            const int ka = ks + (lhi << 4);
            #pragma unroll
            for (int m = 0; m < 2; ++m) {
                const _Float16* ap = &As[(wm * 32 + m * 16 + l15) * KP + ka];
                a[m].u[0] = *(const u32x4*)(ap);        // K +0..7
                a[m].u[1] = *(const u32x4*)(ap + 8);    // K +8..15
            }
            // B frags use the same contiguous k order as A: lanes 0-15 -> K 0..15, lanes 16-31 -> K 16..31 of col N=lane&15
            const int kb = ks + (lhi << 4);
            #pragma unroll
            for (int n = 0; n < 2; ++n) {
                const _Float16* bp = &Bs[(wn * 32 + n * 16 + l15) * KP + kb];
                b[n].u[0] = *(const u32x4*)(bp);        // 16 contiguous halfs
                b[n].u[1] = *(const u32x4*)(bp + 8);
            }
            #pragma unroll
            for (int m = 0; m < 2; ++m)
                #pragma unroll
                for (int n = 0; n < 2; ++n)
                    acc[m][n] = __builtin_amdgcn_wmma_f32_16x16x32_f16(
                        false, a[m].h, false, b[n].h,
                        (short)0, acc[m][n], false, false);
        }

        // One barrier per chunk: publishes buf^1 fills and retires reads of buf.
        __syncthreads();
    }

    // ---- Epilogue: C/D layout -> lanes 0-15: M=v, lanes 16-31: M=8+v; undo coeff scale
    #pragma unroll
    for (int m = 0; m < 2; ++m) {
        const int rbase = row0 + wm * 32 + m * 16 + (lhi << 3);
        #pragma unroll
        for (int n = 0; n < 2; ++n) {
            const int col = col0 + wn * 32 + n * 16 + l15;
            #pragma unroll
            for (int v = 0; v < 8; ++v)
                out[(size_t)(rbase + v) * OUT_DIM + col] = acc[m][n][v] * CSCALE_INV;
        }
    }
}

extern "C" void kernel_launch(void* const* d_in, const int* in_sizes, int n_in,
                              void* d_out, int out_size, void* d_ws, size_t ws_size,
                              hipStream_t stream) {
    const float* x      = (const float*)d_in[0];
    const float* coeffs = (const float*)d_in[1];
    float* out          = (float*)d_out;
    const int Bsz = in_sizes[0] / IN_DIM;          // 4096
    dim3 grid(OUT_DIM / BN, Bsz / BM);             // 4 x 64 = 256 workgroups
    sinckan_wmma<<<grid, NT, 0, stream>>>(x, coeffs, out);
}
